// DilatedSlidingWindowAttention_31791347925506
// MI455X (gfx1250) — matrix operand order repaired and checked by reading, not compiled, not run on hardware
//
#include <hip/hip_runtime.h>
#include <hip/hip_bf16.h>

// ---------------------------------------------------------------------------
// DilatedSlidingWindowAttention for MI455X (gfx1250, wave32, WMMA)
//
// Sizes: B=2 S=2048 D=512 H=8 HD=64 WIN=16 DIL=2 G=64 L=1984
// Strategy: project-then-gather (gather commutes with per-row linear maps),
// bf16 WMMA GEMMs with fp32 accumulate for all 8 projections, VALU attention.
// ---------------------------------------------------------------------------

#define Bsz 2
#define Ssz 2048
#define Dsz 512
#define Hsz 8
#define HDsz 64
#define WINsz 16
#define Gsz 64
#define Lsz (Ssz - Gsz)          // 1984
#define MLOC (Bsz * Lsz)         // 3968  (multiple of 64)
#define MGLB (Bsz * Gsz)         // 128   (multiple of 64)

#define TM 64
#define TN 64
#define TK 32
#define LDS_PAD 8                // 40 bf16 = 80B row stride -> conflict-free, 16B aligned

typedef __attribute__((ext_vector_type(16))) __bf16 v16bf;
typedef __attribute__((ext_vector_type(8)))  float  v8f;

// C[M,N] = A[M,K] @ W[N,K]^T + bias[N]
// A row r lives at  A + (r / a_rpb) * a_bstride + (r % a_rpb) * K
// Out row m lives at Out + (m / o_rpb) * o_bstride + (m % o_rpb) * N
// M % 64 == 0, N % 64 == 0, K % 32 == 0 (guaranteed by caller).
__global__ __launch_bounds__(128) void wmma_gemm_bias(
    const float* __restrict__ A, int a_rpb, long long a_bstride,
    const float* __restrict__ W, const float* __restrict__ bias,
    float* __restrict__ Out, int o_rpb, long long o_bstride,
    int M, int N, int K)
{
    __shared__ __bf16 As[TM][TK + LDS_PAD];
    __shared__ __bf16 Bs[TN][TK + LDS_PAD];

    const int tid  = threadIdx.x;
    const int wave = tid >> 5;
    const int lane = tid & 31;
    const int row0 = blockIdx.x * TM;
    const int col0 = blockIdx.y * TN;

    // staging assignment: each thread converts 16 floats of A and 16 of W
    const int lm = tid >> 1;          // 0..63 : row within tile
    const int lk = (tid & 1) * 16;    // 0 or 16 : k-chunk within tile

    const int r = row0 + lm;
    const float* arow = A + (long long)(r / a_rpb) * a_bstride
                          + (long long)(r % a_rpb) * K + lk;
    const float* wrow = W + (long long)(col0 + lm) * K + lk;

    v8f acc[4] = {v8f{}, v8f{}, v8f{}, v8f{}};

    const int half = lane >> 4;       // which 16-lane half
    const int l16  = lane & 15;
    const int mrow = wave * 16 + l16; // A-fragment row for this wave

    for (int k0 = 0; k0 < K; k0 += TK) {
        // ---- stage fp32 -> bf16 tiles into LDS ----
        #pragma unroll
        for (int i = 0; i < 16; i += 4) {
            const float4 av = *(const float4*)(arow + k0 + i);
            As[lm][lk + i + 0] = (__bf16)av.x;
            As[lm][lk + i + 1] = (__bf16)av.y;
            As[lm][lk + i + 2] = (__bf16)av.z;
            As[lm][lk + i + 3] = (__bf16)av.w;
            const float4 wv = *(const float4*)(wrow + k0 + i);
            Bs[lm][lk + i + 0] = (__bf16)wv.x;
            Bs[lm][lk + i + 1] = (__bf16)wv.y;
            Bs[lm][lk + i + 2] = (__bf16)wv.z;
            Bs[lm][lk + i + 3] = (__bf16)wv.w;
        }
        __syncthreads();

        // ---- A fragment (ISA 16-bit A 16x32 layout) ----
        // elements 0..7  -> K = 8*half + 0..7
        // elements 8..15 -> K = 16 + 8*half + 0..7
        v16bf afrag;
        #pragma unroll
        for (int i = 0; i < 8; ++i) {
            afrag[i]     = As[mrow][8 * half + i];
            afrag[8 + i] = As[mrow][16 + 8 * half + i];
        }

        // ---- 4 N-subtiles: B fragment K = 8*half + 0..7, 16 + 8*half + 0..7, N = lane&15 ----
        #pragma unroll
        for (int nt = 0; nt < 4; ++nt) {
            const int ncol = nt * 16 + l16;
            v16bf bfrag;
            #pragma unroll
            for (int i = 0; i < 16; ++i)
                bfrag[i] = Bs[ncol][(i < 8) ? (8 * half + i) : (16 + 8 * half + (i - 8))];
            acc[nt] = __builtin_amdgcn_wmma_f32_16x16x32_bf16(
                false, afrag, false, bfrag, (short)0, acc[nt], false, false);
        }
        __syncthreads();
    }

    // ---- epilogue: bias add + strided store (C/D row = j + 8*half) ----
    #pragma unroll
    for (int nt = 0; nt < 4; ++nt) {
        const int col = col0 + nt * 16 + l16;
        const float bv = bias[col];
        #pragma unroll
        for (int j = 0; j < 8; ++j) {
            const int m = row0 + wave * 16 + j + 8 * half;
            float* orow = Out + (long long)(m / o_rpb) * o_bstride
                              + (long long)(m % o_rpb) * N;
            orow[col] = acc[nt][j] + bv;
        }
    }
}

// ---------------------------------------------------------------------------
// Local dilated-window attention: one wave per (b, pos, h).
// qp/kp/vp/op are (B*L, D) fp32 with row r = b*L + pos, col = h*64 + d.
// ---------------------------------------------------------------------------
__global__ __launch_bounds__(256) void local_attn(
    const float* __restrict__ qp, const float* __restrict__ kp,
    const float* __restrict__ vp, float* __restrict__ op)
{
    const int task = blockIdx.x * 8 + (threadIdx.x >> 5); // (b*L+pos)*H + h
    const int lane = threadIdx.x & 31;
    const int h = task & (Hsz - 1);
    const int r = task >> 3;                 // b*L + pos
    const int b = r / Lsz;
    const int pos = r - b * Lsz;

    const int w = lane & 15;                 // window tap
    const int half = lane >> 4;              // which 32-dim half of HD=64
    int wp = pos + 2 * (w - 8);
    wp = min(max(wp, 0), Lsz - 1);

    const float* qrow = qp + (long long)r * Dsz + h * HDsz + half * 32;
    const float* krow = kp + (long long)(b * Lsz + wp) * Dsz + h * HDsz + half * 32;

    float s = 0.f;
    #pragma unroll
    for (int d = 0; d < 32; ++d) s += qrow[d] * krow[d];
    s += __shfl_xor(s, 16, 32);              // combine halves -> full 64-dot
    s *= 0.125f;                             // 1/sqrt(64)

    // softmax across the 16 taps (lanes 0..15 and 16..31 hold duplicates)
    float m = s;
    #pragma unroll
    for (int off = 8; off > 0; off >>= 1) m = fmaxf(m, __shfl_xor(m, off, 32));
    float e = __expf(s - m);
    float den = e;
    #pragma unroll
    for (int off = 8; off > 0; off >>= 1) den += __shfl_xor(den, off, 32);
    const float p = e / den;

    // o[d0..d0+1], d0 = 2*lane : gather p from tap lanes, fma with v rows
    float acc0 = 0.f, acc1 = 0.f;
    #pragma unroll
    for (int wi = 0; wi < 16; ++wi) {
        const float pw = __shfl(p, wi, 32);
        int wpi = pos + 2 * (wi - 8);
        wpi = min(max(wpi, 0), Lsz - 1);
        const float2 vv = *(const float2*)(
            vp + (long long)(b * Lsz + wpi) * Dsz + h * HDsz + 2 * lane);
        acc0 += pw * vv.x;
        acc1 += pw * vv.y;
    }
    *(float2*)(op + (long long)r * Dsz + h * HDsz + 2 * lane) =
        make_float2(acc0, acc1);
}

// ---------------------------------------------------------------------------
// Global attention: one block per (b,h), thread = query row (G=64 threads).
// qg/kg/vg/og are (B*G, D) fp32, col = h*64 + d.
// ---------------------------------------------------------------------------
__global__ __launch_bounds__(64) void global_attn(
    const float* __restrict__ qg, const float* __restrict__ kg,
    const float* __restrict__ vg, float* __restrict__ og)
{
    const int b = blockIdx.x >> 3;
    const int h = blockIdx.x & 7;
    const int qi = threadIdx.x;

    float q[HDsz];
    const float* qrow = qg + (long long)(b * Gsz + qi) * Dsz + h * HDsz;
    #pragma unroll
    for (int d = 0; d < HDsz; ++d) q[d] = qrow[d];

    __shared__ float sP[Gsz][Gsz + 1];

    float m = -1e30f;
    for (int k = 0; k < Gsz; ++k) {
        const float* krow = kg + (long long)(b * Gsz + k) * Dsz + h * HDsz;
        float s = 0.f;
        #pragma unroll
        for (int d = 0; d < HDsz; ++d) s += q[d] * krow[d];
        s *= 0.125f;
        sP[qi][k] = s;
        m = fmaxf(m, s);
    }
    float den = 0.f;
    for (int k = 0; k < Gsz; ++k) {
        const float e = __expf(sP[qi][k] - m);
        sP[qi][k] = e;
        den += e;
    }
    const float inv = 1.f / den;

    float acc[HDsz];
    #pragma unroll
    for (int d = 0; d < HDsz; ++d) acc[d] = 0.f;
    for (int k = 0; k < Gsz; ++k) {
        const float p = sP[qi][k] * inv;
        const float* vrow = vg + (long long)(b * Gsz + k) * Dsz + h * HDsz;
        #pragma unroll
        for (int d = 0; d < HDsz; ++d) acc[d] += p * vrow[d];
    }
    float* orow = og + (long long)(b * Gsz + qi) * Dsz + h * HDsz;
    #pragma unroll
    for (int d = 0; d < HDsz; ++d) orow[d] = acc[d];
}

// ---------------------------------------------------------------------------
extern "C" void kernel_launch(void* const* d_in, const int* in_sizes, int n_in,
                              void* d_out, int out_size, void* d_ws, size_t ws_size,
                              hipStream_t stream) {
    const float* query   = (const float*)d_in[0];
    const float* key     = (const float*)d_in[1];
    const float* value   = (const float*)d_in[2];
    const float* wq      = (const float*)d_in[3];
    const float* bq      = (const float*)d_in[4];
    const float* wk      = (const float*)d_in[5];
    const float* bk      = (const float*)d_in[6];
    const float* wv      = (const float*)d_in[7];
    const float* bv      = (const float*)d_in[8];
    const float* wo      = (const float*)d_in[9];
    const float* bo      = (const float*)d_in[10];
    const float* g_in_w  = (const float*)d_in[11];
    const float* g_in_b  = (const float*)d_in[12];
    const float* g_out_w = (const float*)d_in[13];
    const float* g_out_b = (const float*)d_in[14];
    float* out = (float*)d_out;

    // workspace layout (fp32)
    float* f = (float*)d_ws;
    const long long NL = (long long)MLOC * Dsz;   // 3968*512
    const long long NG = (long long)MGLB * Dsz;   // 128*512
    float* ws_q  = f;
    float* ws_k  = f + NL;
    float* ws_v  = f + 2 * NL;
    float* ws_o  = f + 3 * NL;
    float* ws_qg = f + 4 * NL;
    float* ws_kg = ws_qg + NG;
    float* ws_vg = ws_kg + NG;
    float* ws_og = ws_vg + NG;

    const long long bs = (long long)Ssz * Dsz;    // batch stride in input/output
    const dim3 gL(MLOC / TM, Dsz / TN);           // (62, 8)
    const dim3 gG(MGLB / TM, Dsz / TN);           // (2, 8)

    // --- local q/k/v projections (project first, gather later) ---
    wmma_gemm_bias<<<gL, 128, 0, stream>>>(query + Gsz * Dsz, Lsz, bs, wq, bq,
                                           ws_q, MLOC, 0, MLOC, Dsz, Dsz);
    wmma_gemm_bias<<<gL, 128, 0, stream>>>(key + Gsz * Dsz, Lsz, bs, wk, bk,
                                           ws_k, MLOC, 0, MLOC, Dsz, Dsz);
    wmma_gemm_bias<<<gL, 128, 0, stream>>>(value + Gsz * Dsz, Lsz, bs, wv, bv,
                                           ws_v, MLOC, 0, MLOC, Dsz, Dsz);

    // --- local dilated-window attention ---
    local_attn<<<(MLOC * Hsz) / 8, 256, 0, stream>>>(ws_q, ws_k, ws_v, ws_o);

    // --- local output projection, strided straight into out[:, G:, :] ---
    wmma_gemm_bias<<<gL, 128, 0, stream>>>(ws_o, MLOC, 0, wo, bo,
                                           out + Gsz * Dsz, Lsz, bs,
                                           MLOC, Dsz, Dsz);

    // --- global q/k/v projections from query/key/value[:, :G, :] ---
    wmma_gemm_bias<<<gG, 128, 0, stream>>>(query, Gsz, bs, g_in_w, g_in_b,
                                           ws_qg, MGLB, 0, MGLB, Dsz, Dsz);
    wmma_gemm_bias<<<gG, 128, 0, stream>>>(key, Gsz, bs, g_in_w + Dsz * Dsz,
                                           g_in_b + Dsz, ws_kg, MGLB, 0,
                                           MGLB, Dsz, Dsz);
    wmma_gemm_bias<<<gG, 128, 0, stream>>>(value, Gsz, bs, g_in_w + 2 * Dsz * Dsz,
                                           g_in_b + 2 * Dsz, ws_vg, MGLB, 0,
                                           MGLB, Dsz, Dsz);

    // --- global dense attention ---
    global_attn<<<Bsz * Hsz, 64, 0, stream>>>(ws_qg, ws_kg, ws_vg, ws_og);

    // --- global output projection, strided into out[:, :G, :] ---
    wmma_gemm_bias<<<gG, 128, 0, stream>>>(ws_og, MGLB, 0, g_out_w, g_out_b,
                                           out, Gsz, bs, MGLB, Dsz, Dsz);
}
